// VGG16Head_824633721071
// MI455X (gfx1250) — matrix operand order repaired and checked by reading, not compiled, not run on hardware
//
#include <hip/hip_runtime.h>
#include <hip/hip_bf16.h>

#define ROI 7
#define RSCALE (1.0f/16.0f)
#define CB 4
#define CC 512
#define CH 38
#define CW 38
#define CR 512
#define D_IN 25088   // 512*49
#define D_H  4096
#define N_LOCS 84
#define N_SCORES 21

typedef __attribute__((ext_vector_type(16))) __bf16 v16bf;
typedef __attribute__((ext_vector_type(2)))  __bf16 v2bf;
typedef __attribute__((ext_vector_type(8)))  float  v8f;

// ---- bf16 helpers (round-to-nearest-even) -------------------------------
__device__ __forceinline__ unsigned short f2bf(float f) {
    unsigned u = __builtin_bit_cast(unsigned, f);
    return (unsigned short)((u + 0x7FFFu + ((u >> 16) & 1u)) >> 16);
}
__device__ __forceinline__ unsigned pack_bf16(float lo, float hi) {
#if __has_builtin(__builtin_amdgcn_cvt_pk_bf16_f32)
    v2bf r = __builtin_amdgcn_cvt_pk_bf16_f32(lo, hi);
    return __builtin_bit_cast(unsigned, r);
#else
    unsigned a = __builtin_bit_cast(unsigned, lo);
    unsigned b = __builtin_bit_cast(unsigned, hi);
    unsigned ra = (a + 0x7FFFu + ((a >> 16) & 1u)) >> 16;
    unsigned rb = (b + 0x7FFFu + ((b >> 16) & 1u)) & 0xFFFF0000u;
    return ra | rb;
#endif
}
__device__ __forceinline__ uint4 pack4(float4 lo, float4 hi) {
    uint4 p;
    p.x = pack_bf16(lo.x, hi.x);
    p.y = pack_bf16(lo.y, hi.y);
    p.z = pack_bf16(lo.z, hi.z);
    p.w = pack_bf16(lo.w, hi.w);
    return p;
}

// CDNA5 async global->LDS copy (ASYNCcnt-tracked).  dst = LDS byte offset,
// src = 64-bit global address.  16 bytes per lane.
__device__ __forceinline__ void async_load_b128(unsigned lds_off, const void* gaddr) {
    asm volatile("global_load_async_to_lds_b128 %0, %1, off"
                 :: "v"(lds_off), "v"(gaddr) : "memory");
}
__device__ __forceinline__ void wait_asynccnt0() {
    asm volatile("s_wait_asynccnt 0x0" ::: "memory");
}

// =========================================================================
// Kernel 1: ROI max pooling -> bf16 activations [R, C*49] (GEMM A layout)
// =========================================================================
__global__ __launch_bounds__(256)
void roi_pool_kernel(const float* __restrict__ fmaps,
                     const float* __restrict__ boxes,
                     const int*   __restrict__ binds,
                     unsigned short* __restrict__ pooled) {
    int idx = blockIdx.x * 256 + threadIdx.x;
    const int total = CR * CC * ROI * ROI;
    if (idx >= total) return;
    int pw = idx % ROI;
    int ph = (idx / ROI) % ROI;
    int c  = (idx / (ROI * ROI)) % CC;
    int r  = idx / (ROI * ROI * CC);

    float y1 = boxes[r * 4 + 0], x1 = boxes[r * 4 + 1];
    float y2 = boxes[r * 4 + 2], x2 = boxes[r * 4 + 3];
    float ys = rintf(y1 * RSCALE), xs = rintf(x1 * RSCALE);
    float ye = rintf(y2 * RSCALE), xe = rintf(x2 * RSCALE);
    float rh = fmaxf(ye - ys + 1.0f, 1.0f);
    float rw = fmaxf(xe - xs + 1.0f, 1.0f);

    float hs = fminf(fmaxf(ys + floorf((float)ph * rh / 7.0f), 0.0f), (float)CH);
    float he = fminf(fmaxf(ys + ceilf((float)(ph + 1) * rh / 7.0f), 0.0f), (float)CH);
    float wsf = fminf(fmaxf(xs + floorf((float)pw * rw / 7.0f), 0.0f), (float)CW);
    float wef = fminf(fmaxf(xs + ceilf((float)(pw + 1) * rw / 7.0f), 0.0f), (float)CW);
    int h0 = (int)hs, h1 = (int)he;
    int w0 = (int)wsf, w1 = (int)wef;

    int b = binds[r];
    const float* f = fmaps + ((size_t)b * CC + c) * (CH * CW);
    float m = -1e30f;
    for (int h = h0; h < h1; ++h)
        for (int w = w0; w < w1; ++w)
            m = fmaxf(m, f[h * CW + w]);
    if (m <= -5e29f) m = 0.0f;   // empty bin convention

    pooled[(size_t)r * D_IN + (c * 49 + ph * 7 + pw)] = f2bf(m);
}

// =========================================================================
// Kernel 2: bf16 WMMA GEMM, double-buffered, async A staging.
// A: bf16 [M,K] row-major (ws). B: fp32 [K,N] row-major weights, converted
// to packed-bf16 pairs on the fly. Out: bf16 [M,N], bias + optional ReLU.
// BM=128 BN=128 BK=32; 8 waves x (2 M-tiles x 4 N-tiles).
// =========================================================================
#define BM 128
#define BN 128
#define BK 32
#define ASTR 20    // u32 stride per A row (16 pairs + pad; 16B aligned; conflict-free)
#define BSTR 132   // u32 stride per B kp row (128 + pad)

__global__ __launch_bounds__(256)
void gemm_bf16_kernel(const unsigned short* __restrict__ A,
                      const float* __restrict__ Bw,
                      const float* __restrict__ bias,
                      unsigned short* __restrict__ Out,
                      int M, int N, int K, int relu) {
    __shared__ __align__(16) unsigned As[2][BM * ASTR];        // 2 x 10240 B
    __shared__ __align__(16) unsigned Bs[2][(BK / 2) * BSTR];  // 2 x 8448 B

    const int tid  = threadIdx.x;
    const int wave = tid >> 5;
    const int lane = tid & 31;
    const int l15  = lane & 15;
    const int hsel = lane >> 4;              // 0: lanes 0-15, 1: lanes 16-31
    const int m0 = blockIdx.y * BM;
    const int n0 = blockIdx.x * BN;
    const int wm = (wave >> 1) * 2;          // first of 2 M-tiles
    const int wn = (wave & 1) * 4;           // first of 4 N-tiles

    // per-thread staging descriptors
    const int arow = tid >> 1, ahalf = tid & 1;
    const unsigned short* aptr = A + (size_t)(m0 + arow) * K + ahalf * 16;
    const unsigned lds_a0 = (unsigned)(size_t)&As[0][arow * ASTR + ahalf * 8];
    const unsigned lds_a1 = (unsigned)(size_t)&As[1][arow * ASTR + ahalf * 8];

    const int bn4  = (tid & 31) * 4;
    const int bkp0 = tid >> 5;               // 0..7
    const int bkp1 = bkp0 + 8;               // 8..15
    const float* bptr0 = Bw + (size_t)(2 * bkp0) * N + n0 + bn4;
    const float* bptr1 = Bw + (size_t)(2 * bkp1) * N + n0 + bn4;

    v8f acc[2][4] = {};

    // ---- prologue: stage step 0 into buffer 0
    {
        async_load_b128(lds_a0, aptr);
        async_load_b128(lds_a0 + 16, aptr + 8);
        float4 lo0 = *(const float4*)bptr0, hi0 = *(const float4*)(bptr0 + N);
        float4 lo1 = *(const float4*)bptr1, hi1 = *(const float4*)(bptr1 + N);
        *(uint4*)&Bs[0][bkp0 * BSTR + bn4] = pack4(lo0, hi0);
        *(uint4*)&Bs[0][bkp1 * BSTR + bn4] = pack4(lo1, hi1);
        wait_asynccnt0();
        __syncthreads();
    }

    const int nsteps = K / BK;
    for (int s = 0; s < nsteps; ++s) {
        const int cur = s & 1;
        const unsigned* as  = As[cur];
        const unsigned* bsr = Bs[cur];
        const bool pf = (s + 1 < nsteps);

        // ---- prefetch step s+1: async A -> LDS[next], B fp32 -> registers
        float4 lo0, hi0, lo1, hi1;
        if (pf) {
            const unsigned short* ap = aptr + (size_t)(s + 1) * BK;
            unsigned la = cur ? lds_a0 : lds_a1;
            async_load_b128(la, ap);
            async_load_b128(la + 16, ap + 8);
            const float* p0 = bptr0 + (size_t)(s + 1) * BK * N;
            const float* p1 = bptr1 + (size_t)(s + 1) * BK * N;
            lo0 = *(const float4*)p0; hi0 = *(const float4*)(p0 + N);
            lo1 = *(const float4*)p1; hi1 = *(const float4*)(p1 + N);
        }

        // ---- load fragments (ISA 7.12.2 16-bit A/B layouts, wave32)
        union { v16bf v; unsigned u[8]; } af[2], bf[4];
        #pragma unroll
        for (int i = 0; i < 2; ++i) {
            int base = ((wm + i) * 16 + l15) * ASTR + hsel * 4;
            #pragma unroll
            for (int p = 0; p < 4; ++p) af[i].u[p]     = as[base + p];
            #pragma unroll
            for (int p = 0; p < 4; ++p) af[i].u[4 + p] = as[base + 8 + p];
        }
        #pragma unroll
        for (int j = 0; j < 4; ++j) {
            int col = (wn + j) * 16 + l15;
            int kb  = hsel * 4;
            #pragma unroll
            for (int p = 0; p < 8; ++p)
                bf[j].u[p] = bsr[(kb + p + (p >= 4 ? 4 : 0)) * BSTR + col];
        }

        // ---- 8 WMMAs per wave per K-step
        #pragma unroll
        for (int i = 0; i < 2; ++i)
            #pragma unroll
            for (int j = 0; j < 4; ++j)
                acc[i][j] = __builtin_amdgcn_wmma_f32_16x16x32_bf16(
                    false, af[i].v, false, bf[j].v,
                    (short)0, acc[i][j], false, false);

        // ---- convert + store prefetched B into next buffer
        if (pf) {
            unsigned* bsw = Bs[cur ^ 1];
            *(uint4*)&bsw[bkp0 * BSTR + bn4] = pack4(lo0, hi0);
            *(uint4*)&bsw[bkp1 * BSTR + bn4] = pack4(lo1, hi1);
        }

        wait_asynccnt0();
        __syncthreads();
    }

    // ---- epilogue: bias + ReLU, store bf16 (C/D layout: vgpr e -> M=e+8*hsel, N=l15)
    #pragma unroll
    for (int i = 0; i < 2; ++i) {
        #pragma unroll
        for (int j = 0; j < 4; ++j) {
            int ng = n0 + (wn + j) * 16 + l15;
            float bv = bias[ng];
            #pragma unroll
            for (int e = 0; e < 8; ++e) {
                int mg = m0 + (wm + i) * 16 + e + hsel * 8;
                float v = acc[i][j][e] + bv;
                if (relu) v = fmaxf(v, 0.0f);
                Out[(size_t)mg * N + ng] = f2bf(v);
            }
        }
    }
}

// =========================================================================
// Kernel 3: head GEMMs (N=84 and N=21), fp32 out concatenated in d_out.
// One block per ROI row; fc7 row staged to LDS.
// =========================================================================
__global__ __launch_bounds__(128)
void heads_kernel(const unsigned short* __restrict__ fc7,
                  const float* __restrict__ Wl, const float* __restrict__ bl,
                  const float* __restrict__ Wsc, const float* __restrict__ bsc,
                  float* __restrict__ out) {
    __shared__ float row[D_H];
    int r = blockIdx.x;
    for (int k = threadIdx.x; k < D_H; k += 128) {
        unsigned u = ((unsigned)fc7[(size_t)r * D_H + k]) << 16;
        row[k] = __builtin_bit_cast(float, u);
    }
    __syncthreads();
    int j = threadIdx.x;
    if (j < N_LOCS) {
        float acc = bl[j];
        for (int k = 0; k < D_H; ++k) acc += row[k] * Wl[(size_t)k * N_LOCS + j];
        out[(size_t)r * N_LOCS + j] = acc;
    } else if (j < N_LOCS + N_SCORES) {
        int jj = j - N_LOCS;
        float acc = bsc[jj];
        for (int k = 0; k < D_H; ++k) acc += row[k] * Wsc[(size_t)k * N_SCORES + jj];
        out[(size_t)CR * N_LOCS + (size_t)r * N_SCORES + jj] = acc;
    }
}

// =========================================================================
extern "C" void kernel_launch(void* const* d_in, const int* in_sizes, int n_in,
                              void* d_out, int out_size, void* d_ws, size_t ws_size,
                              hipStream_t stream) {
    const float* fmaps = (const float*)d_in[0];
    const float* boxes = (const float*)d_in[1];
    const int*   binds = (const int*)d_in[2];
    const float* Wfc6  = (const float*)d_in[3];
    const float* bfc6  = (const float*)d_in[4];
    const float* Wfc7  = (const float*)d_in[5];
    const float* bfc7  = (const float*)d_in[6];
    const float* Wl    = (const float*)d_in[7];
    const float* bl    = (const float*)d_in[8];
    const float* Wsc   = (const float*)d_in[9];
    const float* bsc   = (const float*)d_in[10];
    float* out = (float*)d_out;

    char* ws = (char*)d_ws;
    unsigned short* pooled = (unsigned short*)ws;                                  // 25.7 MB bf16
    unsigned short* fc6    = (unsigned short*)(ws + (size_t)CR * D_IN * 2);        // 4 MB bf16
    unsigned short* fc7    = (unsigned short*)(ws + (size_t)CR * D_IN * 2 + (size_t)CR * D_H * 2);

    // 1) ROI pool -> bf16 A matrix
    {
        int total  = CR * CC * ROI * ROI;              // 12,845,056
        int blocks = (total + 255) / 256;
        roi_pool_kernel<<<blocks, 256, 0, stream>>>(fmaps, boxes, binds, pooled);
    }
    // 2) fc6: [512,25088] x [25088,4096] + relu
    dim3 g(D_H / BN, CR / BM);   // (32, 4)
    gemm_bf16_kernel<<<g, 256, 0, stream>>>(pooled, Wfc6, bfc6, fc6, CR, D_H, D_IN, 1);
    // 3) fc7: [512,4096] x [4096,4096] + relu
    gemm_bf16_kernel<<<g, 256, 0, stream>>>(fc6, Wfc7, bfc7, fc7, CR, D_H, D_H, 1);
    // 4) heads -> d_out (locs [512*84] then scores [512*21])
    heads_kernel<<<CR, 128, 0, stream>>>(fc7, Wl, bl, Wsc, bsc, out);
}
